// LightweightCNNGeometricMorphism_35820027249052
// MI455X (gfx1250) — matrix operand order repaired and checked by reading, not compiled, not run on hardware
//
#include <hip/hip_runtime.h>
#include <hip/hip_bf16.h>

// ---------------------------------------------------------------------------
// Problem dims
#define BATCH 8
#define CH    64
#define HHH   64
#define WWW   64
#define NTOK  (HHH * WWW)      // 4096 tokens per batch
#define OUT_H 32
#define OUT_W 32
#define LOG2E 1.4426950408889634f

typedef __attribute__((ext_vector_type(16))) __bf16 v16bf;
typedef __attribute__((ext_vector_type(8)))  float  v8f;

#define AS1 __attribute__((address_space(1)))
#define AS3 __attribute__((address_space(3)))

typedef int b128_t __attribute__((vector_size(16)));   // matches builtin param type

union AFrag {
    v16bf v;
    unsigned int w[8];
};

static __device__ __forceinline__ unsigned short f32_to_bf16(float f) {
    unsigned int u = __float_as_uint(f);
    unsigned int r = u + 0x7FFFu + ((u >> 16) & 1u);   // round-to-nearest-even
    return (unsigned short)(r >> 16);
}

static __device__ __forceinline__ v8f zero8() {
    v8f z;
#pragma unroll
    for (int i = 0; i < 8; ++i) z[i] = 0.0f;
    return z;
}

// --- CDNA5 async global->LDS staging (ASYNCcnt-tracked) --------------------
#if __has_builtin(__builtin_amdgcn_global_load_async_to_lds_b128)
#define HAVE_ASYNC_LDS 1
#else
#define HAVE_ASYNC_LDS 0
#endif

static __device__ __forceinline__ AS3 b128_t* to_lds(void* p) {
    // flat LDS address: low 32 bits are the LDS byte offset (ISA 10.2)
    return (AS3 b128_t*)(unsigned int)(unsigned long long)p;
}
static __device__ __forceinline__ AS1 b128_t* to_glob(const void* p) {
    return (AS1 b128_t*)(unsigned long long)p;
}

static __device__ __forceinline__ void stage_b128(const void* g, void* l) {
#if HAVE_ASYNC_LDS
    __builtin_amdgcn_global_load_async_to_lds_b128(to_glob(g), to_lds(l), 0, 0);
#else
    *(uint4*)l = *(const uint4*)g;   // sync fallback (same barrier structure)
#endif
}

static __device__ __forceinline__ void wait_async_le8() {
#if HAVE_ASYNC_LDS
#if __has_builtin(__builtin_amdgcn_s_wait_asynccnt)
    __builtin_amdgcn_s_wait_asynccnt(8);
#else
    asm volatile("s_wait_asynccnt 0x8" ::: "memory");
#endif
#endif
}
static __device__ __forceinline__ void wait_async_le0() {
#if HAVE_ASYNC_LDS
#if __has_builtin(__builtin_amdgcn_s_wait_asynccnt)
    __builtin_amdgcn_s_wait_asynccnt(0);
#else
    asm volatile("s_wait_asynccnt 0x0" ::: "memory");
#endif
#endif
}

// A-matrix fragment (16x32 bf16): lane%16 = M row; half (lane/16) selects K
// octet.  Per ISA VGPR table: v in 0..3 -> K base (v*2), v in 4..7 -> 16+...
static __device__ __forceinline__ void load_a_frag(const unsigned short* buf,
                                                   int rowlen, int row, int k0,
                                                   int hf, AFrag& f) {
    const unsigned int* b32 = (const unsigned int*)buf;
#pragma unroll
    for (int v = 0; v < 8; ++v) {
        int kb = k0 + ((v & 4) << 2) + ((v & 3) << 1) + hf * 8;
        f.w[v] = b32[(row * rowlen + kb) >> 1];
    }
}

// B-matrix fragment (32x16 bf16): lane%16 = N col; half selects K octet (ISA table order).
// buf is stored [col][k] so each pair (k,k+1) is one aligned dword.
static __device__ __forceinline__ void load_b_frag(const unsigned short* buf,
                                                   int rowlen, int col, int k0,
                                                   int hf, AFrag& f) {
    const unsigned int* b32 = (const unsigned int*)buf;
#pragma unroll
    for (int v = 0; v < 8; ++v) {
        int kb = k0 + ((v & 4) << 2) + ((v & 3) << 1) + hf * 8;
        f.w[v] = b32[(col * rowlen + kb) >> 1];
    }
}

static __device__ __forceinline__ v8f wmma_bf16(const AFrag& a, const AFrag& b, v8f c) {
    return __builtin_amdgcn_wmma_f32_16x16x32_bf16(
        /*neg_a=*/false, a.v, /*neg_b=*/false, b.v,
        /*c_mod=*/(short)0, c, /*reuse_a=*/false, /*reuse_b=*/false);
}

static __device__ __forceinline__ float red_max16(float x) {
#pragma unroll
    for (int m = 1; m <= 8; m <<= 1) x = fmaxf(x, __shfl_xor(x, m, 32));
    return x;  // reduction confined to 16-lane half (masks <= 8)
}
static __device__ __forceinline__ float red_sum16(float x) {
#pragma unroll
    for (int m = 1; m <= 8; m <<= 1) x += __shfl_xor(x, m, 32);
    return x;
}

// ---------------------------------------------------------------------------
// Kernel 1: Q/K/V projections.  Q,K stored bf16 [b][tok][c]; V stored bf16
// transposed [b][c][tok].  Q pre-scaled by 1/sqrt(C)=0.125 (exact in bf16).
__global__ void qkv_kernel(const float* __restrict__ x,
                           const float* __restrict__ wq, const float* __restrict__ bq,
                           const float* __restrict__ wk, const float* __restrict__ bk,
                           const float* __restrict__ wv, const float* __restrict__ bv,
                           unsigned short* __restrict__ Q,
                           unsigned short* __restrict__ K,
                           unsigned short* __restrict__ Vt) {
    __shared__ __align__(16) unsigned short t_lds[CH * CH];    // token tile [n][c]
    __shared__ __align__(16) unsigned short wq_lds[CH * CH];   // [d][c]
    __shared__ __align__(16) unsigned short wk_lds[CH * CH];
    __shared__ __align__(16) unsigned short wv_lds[CH * CH];

    const int tid  = threadIdx.x;
    const int b    = blockIdx.y;
    const int tok0 = blockIdx.x * 64;

    // Load token tile: x[b][c][tok0+n] -> t_lds[n][c] (bf16).  Coalesced over n.
#pragma unroll 4
    for (int i = 0; i < 32; ++i) {
        int e = i * 128 + tid;
        int c = e >> 6, n = e & 63;
        t_lds[n * 64 + c] = f32_to_bf16(x[((size_t)b * CH + c) * NTOK + tok0 + n]);
    }
#pragma unroll 4
    for (int i = 0; i < 32; ++i) {
        int e = i * 128 + tid;
        wq_lds[e] = f32_to_bf16(wq[e]);
        wk_lds[e] = f32_to_bf16(wk[e]);
        wv_lds[e] = f32_to_bf16(wv[e]);
    }
    __syncthreads();

    const int wave = tid >> 5, lane = tid & 31;
    const int hf = lane >> 4, l = lane & 15;
    const int row0 = wave * 16;

    AFrag a[2];
    load_a_frag(t_lds, 64, row0 + l, 0,  hf, a[0]);
    load_a_frag(t_lds, 64, row0 + l, 32, hf, a[1]);

#pragma unroll
    for (int which = 0; which < 3; ++which) {
        const unsigned short* wlds = (which == 0) ? wq_lds : (which == 1) ? wk_lds : wv_lds;
        const float* bptr          = (which == 0) ? bq     : (which == 1) ? bk     : bv;
#pragma unroll
        for (int df = 0; df < 4; ++df) {
            AFrag b0, b1;
            load_b_frag(wlds, 64, df * 16 + l, 0,  hf, b0);
            load_b_frag(wlds, 64, df * 16 + l, 32, hf, b1);
            v8f acc = zero8();
            acc = wmma_bf16(a[0], b0, acc);
            acc = wmma_bf16(a[1], b1, acc);
            const int d = df * 16 + l;
            const float bias = bptr[d];
#pragma unroll
            for (int vv = 0; vv < 8; ++vv) {
                int token = tok0 + row0 + vv + 8 * hf;
                float val = acc[vv] + bias;
                if (which == 0)
                    Q[((size_t)b * NTOK + token) * CH + d] = f32_to_bf16(val * 0.125f);
                else if (which == 1)
                    K[((size_t)b * NTOK + token) * CH + d] = f32_to_bf16(val);
                else
                    Vt[((size_t)b * CH + d) * NTOK + token] = f32_to_bf16(val);
            }
        }
    }
}

// ---------------------------------------------------------------------------
// Kernel 2: flash attention over 4096 keys (64 tiles of 64), double-buffered
// async K/V staging, fused with the output mix GEMM + bias + ReLU.
__global__ void attn_kernel(const unsigned short* __restrict__ Q,
                            const unsigned short* __restrict__ K,
                            const unsigned short* __restrict__ Vt,
                            const float* __restrict__ wm,
                            const float* __restrict__ bm,
                            float* __restrict__ mixed) {
    __shared__ __align__(16) unsigned short k_lds[2][64 * 64];  // [buf][keyN][c]
    __shared__ __align__(16) unsigned short v_lds[2][64 * 64];  // [buf][c][keyN]
    __shared__ __align__(16) unsigned short wm_lds[CH * CH];    // [d][c]
    __shared__ __align__(16) unsigned short p_lds[4][16 * 64];  // per-wave P / O tile

    const int tid  = threadIdx.x;
    const int b    = blockIdx.y;
    const int tok0 = blockIdx.x * 64;
    const int wave = tid >> 5, lane = tid & 31;
    const int hf = lane >> 4, l = lane & 15;
    const int row0 = wave * 16;

    const char* kbase0 = (const char*)(K + (size_t)b * NTOK * CH);       // +j*8192
    const char* vbase0 = (const char*)(Vt + (size_t)b * CH * NTOK);      // rows stride 8192B

    // Stage one 64-key tile (8 KB K + 8 KB V) into LDS buffer `buf`.
    // Per thread: 4+4 b128 async transfers (whole block covers the tile).
    auto issue_tile = [&](int j, int buf) {
        const char* kb = kbase0 + (size_t)j * (64 * CH * 2);
        char* kd = (char*)k_lds[buf];
#pragma unroll
        for (int i = 0; i < 4; ++i) {
            int e = i * 128 + tid;                       // 16B chunk id, 0..511
            stage_b128(kb + e * 16, kd + e * 16);
        }
        const char* vb = vbase0 + (size_t)j * (64 * 2);  // tile col offset in row
        char* vd = (char*)v_lds[buf];
#pragma unroll
        for (int i = 0; i < 4; ++i) {
            int e = i * 128 + tid;
            int c = e >> 3, cc = e & 7;                  // row c, 16B chunk cc
            stage_b128(vb + (size_t)c * (NTOK * 2) + cc * 16,
                       vd + c * 128 + cc * 16);
        }
    };

#pragma unroll 4
    for (int i = 0; i < 32; ++i) {
        int e = i * 128 + tid;
        wm_lds[e] = f32_to_bf16(wm[e]);
    }

    // Q A-fragments for this wave's 16 rows (held in registers for the loop)
    AFrag qa[2];
    {
        const unsigned short* qbase = Q + ((size_t)b * NTOK + tok0 + row0) * CH;
        load_a_frag(qbase, 64, l, 0,  hf, qa[0]);
        load_a_frag(qbase, 64, l, 32, hf, qa[1]);
    }

    float rmax[8], rsum[8];
#pragma unroll
    for (int v = 0; v < 8; ++v) { rmax[v] = -3.0e38f; rsum[v] = 0.0f; }
    v8f o[4];
#pragma unroll
    for (int cf = 0; cf < 4; ++cf) o[cf] = zero8();

    issue_tile(0, 0);   // prologue: tile 0 in flight

    for (int j = 0; j < 64; ++j) {
        const int buf = j & 1;
        __syncthreads();             // all waves done reading buf^1 (iter j-1)
        if (j < 63) {
            issue_tile(j + 1, buf ^ 1);
            wait_async_le8();        // tile j complete; only tile j+1 in flight
        } else {
            wait_async_le0();
        }
        __syncthreads();             // tile j visible to every wave

        // --- S = Q K^T (already includes 1/sqrt(C) via pre-scaled Q) ---
        v8f s[4];
#pragma unroll
        for (int nf = 0; nf < 4; ++nf) {
            AFrag b0, b1;
            load_b_frag(k_lds[buf], 64, nf * 16 + l, 0,  hf, b0);
            load_b_frag(k_lds[buf], 64, nf * 16 + l, 32, hf, b1);
            v8f acc = zero8();
            acc = wmma_bf16(qa[0], b0, acc);
            acc = wmma_bf16(qa[1], b1, acc);
            s[nf] = acc;
        }

        // --- online softmax, per owned row (VGPR slot vv, 16-lane half) ---
#pragma unroll
        for (int vv = 0; vv < 8; ++vv) {
            float mx = fmaxf(fmaxf(s[0][vv], s[1][vv]), fmaxf(s[2][vv], s[3][vv]));
            mx = red_max16(mx);
            float nm = fmaxf(rmax[vv], mx);
            float sc = exp2f((rmax[vv] - nm) * LOG2E);
            rmax[vv] = nm;
            float ps = 0.0f;
#pragma unroll
            for (int nf = 0; nf < 4; ++nf) {
                float p = exp2f((s[nf][vv] - nm) * LOG2E);
                s[nf][vv] = p;
                ps += p;
            }
            ps = red_sum16(ps);
            rsum[vv] = rsum[vv] * sc + ps;
#pragma unroll
            for (int cf = 0; cf < 4; ++cf) o[cf][vv] *= sc;
#pragma unroll
            for (int nf = 0; nf < 4; ++nf)
                p_lds[wave][(vv + 8 * hf) * 64 + nf * 16 + l] = f32_to_bf16(s[nf][vv]);
        }
        __syncthreads();   // P visible (D-layout -> A-layout round trip)

        // --- O += P V ---
        AFrag pa[2];
        load_a_frag(p_lds[wave], 64, l, 0,  hf, pa[0]);
        load_a_frag(p_lds[wave], 64, l, 32, hf, pa[1]);
#pragma unroll
        for (int cf = 0; cf < 4; ++cf) {
            AFrag b0, b1;
            load_b_frag(v_lds[buf], 64, cf * 16 + l, 0,  hf, b0);
            load_b_frag(v_lds[buf], 64, cf * 16 + l, 32, hf, b1);
            o[cf] = wmma_bf16(pa[0], b0, o[cf]);
            o[cf] = wmma_bf16(pa[1], b1, o[cf]);
        }
    }

    // --- normalize and park O tile (bf16) in per-wave LDS ---
#pragma unroll
    for (int vv = 0; vv < 8; ++vv) {
        float inv = 1.0f / rsum[vv];
#pragma unroll
        for (int cf = 0; cf < 4; ++cf)
            p_lds[wave][(vv + 8 * hf) * 64 + cf * 16 + l] = f32_to_bf16(o[cf][vv] * inv);
    }
    __syncthreads();

    // --- mixed = relu(O * wm^T + bm), store [b][token][d] fp32 ---
    AFrag oa[2];
    load_a_frag(p_lds[wave], 64, l, 0,  hf, oa[0]);
    load_a_frag(p_lds[wave], 64, l, 32, hf, oa[1]);
#pragma unroll
    for (int df = 0; df < 4; ++df) {
        AFrag b0, b1;
        load_b_frag(wm_lds, 64, df * 16 + l, 0,  hf, b0);
        load_b_frag(wm_lds, 64, df * 16 + l, 32, hf, b1);
        v8f acc = zero8();
        acc = wmma_bf16(oa[0], b0, acc);
        acc = wmma_bf16(oa[1], b1, acc);
        const int d = df * 16 + l;
        const float bias = bm[d];
#pragma unroll
        for (int vv = 0; vv < 8; ++vv) {
            int token = tok0 + row0 + vv + 8 * hf;
            float val = fmaxf(acc[vv] + bias, 0.0f);
            mixed[((size_t)b * NTOK + token) * CH + d] = val;
        }
    }
}

// ---------------------------------------------------------------------------
// Kernel 3: bilinear 2x downsample (== 2x2 average for half-pixel centers).
__global__ void pool_kernel(const float* __restrict__ mixed, float* __restrict__ out) {
    int idx = blockIdx.x * 256 + threadIdx.x;   // [b][d][oh][ow]
    int ow = idx & 31;
    int oh = (idx >> 5) & 31;
    int d  = (idx >> 10) & 63;
    int b  = idx >> 16;
    const float* base = mixed + ((size_t)b * NTOK + (oh * 2) * WWW + ow * 2) * CH + d;
    float s = base[0] + base[CH] + base[WWW * CH] + base[WWW * CH + CH];
    out[idx] = 0.25f * s;
}

// ---------------------------------------------------------------------------
extern "C" void kernel_launch(void* const* d_in, const int* in_sizes, int n_in,
                              void* d_out, int out_size, void* d_ws, size_t ws_size,
                              hipStream_t stream) {
    const float* x  = (const float*)d_in[0];
    const float* wq = (const float*)d_in[1];
    const float* bq = (const float*)d_in[2];
    const float* wk = (const float*)d_in[3];
    const float* bk = (const float*)d_in[4];
    const float* wv = (const float*)d_in[5];
    const float* bv = (const float*)d_in[6];
    const float* wm = (const float*)d_in[7];
    const float* bm = (const float*)d_in[8];

    const size_t nQKV = (size_t)BATCH * NTOK * CH;   // 2,097,152 elements
    unsigned short* Q  = (unsigned short*)d_ws;
    unsigned short* K  = Q + nQKV;
    unsigned short* Vt = K + nQKV;
    float* mixed = (float*)(Vt + nQKV);              // 8 MB fp32, after 12 MB bf16

    qkv_kernel<<<dim3(NTOK / 64, BATCH), 128, 0, stream>>>(x, wq, bq, wk, bk, wv, bv, Q, K, Vt);
    attn_kernel<<<dim3(NTOK / 64, BATCH), 128, 0, stream>>>(Q, K, Vt, wm, bm, mixed);
    pool_kernel<<<(BATCH * CH * OUT_H * OUT_W) / 256, 256, 0, stream>>>(mixed, (float*)d_out);
}
